// SimpleRNNTextRepresentation_74062416052827
// MI455X (gfx1250) — matrix operand order repaired and checked by reading, not compiled, not run on hardware
//
#include <hip/hip_runtime.h>

#define B_   256
#define T_   512
#define EMB_ 128
#define HID_ 256
#define OUT_ 128

typedef __attribute__((ext_vector_type(16))) __bf16       v16bf;
typedef __attribute__((ext_vector_type(8)))  float        v8f;
typedef __attribute__((ext_vector_type(4)))  unsigned int v4u;

union AB32 { v16bf v; v4u u[2]; };

static __device__ __forceinline__ unsigned short f2bf(float f) {
    unsigned int u = __float_as_uint(f);
    u += 0x7FFFu + ((u >> 16) & 1u);          // round-to-nearest-even
    return (unsigned short)(u >> 16);
}
static __device__ __forceinline__ unsigned int pk2(float a, float b) {
    return (unsigned int)f2bf(a) | ((unsigned int)f2bf(b) << 16);
}
static __device__ __forceinline__ v8f wmma_bf16(const AB32& a, const AB32& b, v8f c) {
    return __builtin_amdgcn_wmma_f32_16x16x32_bf16(false, a.v, false, b.v,
                                                   (short)0, c, false, false);
}

// Generic shared pointer -> 32-bit LDS byte offset (low 32 bits of the
// shared-aperture generic address are the LDS offset per the ISA).
static __device__ __forceinline__ unsigned int lds_off32(const void* p) {
    return (unsigned int)(unsigned long long)p;
}

// Issue one 16KB tile copy (global -> LDS) as 4 async b128 instructions per
// thread.  INST_OFFSET is added to both the LDS and the global address, so a
// single base register pair serves all four chunks.  Tracked by ASYNCcnt.
static __device__ __forceinline__ void issue_tile_copy_async(const void* gsrc,
                                                             unsigned int ldst) {
    asm volatile(
        "global_load_async_to_lds_b128 %0, %1, off\n\t"
        "global_load_async_to_lds_b128 %0, %1, off offset:4096\n\t"
        "global_load_async_to_lds_b128 %0, %1, off offset:8192\n\t"
        "global_load_async_to_lds_b128 %0, %1, off offset:12288"
        :: "v"(ldst), "v"(gsrc) : "memory");
}

// ---------------------------------------------------------------------------
// Prep: convert weights to bf16, fold the two RNN biases together.
// ---------------------------------------------------------------------------
__global__ void rnn_prep_kernel(const float* __restrict__ W_ih,
                                const float* __restrict__ W_hh,
                                const float* __restrict__ b_ih,
                                const float* __restrict__ b_hh,
                                const float* __restrict__ Wp,
                                unsigned short* __restrict__ wih_bf,
                                unsigned short* __restrict__ whh_bf,
                                unsigned short* __restrict__ wp_bf,
                                float* __restrict__ bias_sum) {
    int tid = blockIdx.x * blockDim.x + threadIdx.x;
    int stride = gridDim.x * blockDim.x;
    for (int i = tid; i < HID_ * EMB_; i += stride) wih_bf[i] = f2bf(W_ih[i]);
    for (int i = tid; i < HID_ * HID_; i += stride) whh_bf[i] = f2bf(W_hh[i]);
    for (int i = tid; i < OUT_ * HID_; i += stride) wp_bf[i]  = f2bf(Wp[i]);
    for (int i = tid; i < HID_;        i += stride) bias_sum[i] = b_ih[i] + b_hh[i];
}

// ---------------------------------------------------------------------------
// Embedding gather + input GEMM: pre[t][b][h] = emb[x[b,t]] @ W_ih^T + bias.
// One block per (btile of 16 batch rows, timestep t).
// ---------------------------------------------------------------------------
__global__ void __launch_bounds__(256)
embed_gemm_kernel(const int* __restrict__ x,
                  const float* __restrict__ emb,
                  const unsigned short* __restrict__ wih_bf,
                  const float* __restrict__ bias_sum,
                  float* __restrict__ pre) {
    __shared__ int tok[16];
    __shared__ __align__(16) unsigned short a_sh[16][136];  // 16 x 128 bf16, padded

    const int btile = blockIdx.x;
    const int t     = blockIdx.y;
    const int tid   = threadIdx.x;

    if (tid < 16) tok[tid] = x[(btile * 16 + tid) * T_ + t];
    __syncthreads();

    // Cooperative gather of 16 embedding rows, f32 -> bf16 into LDS.
    {
        int row = tid >> 4, cg = tid & 15;                 // 8 f32 per thread
        const float* erow = emb + (size_t)tok[row] * EMB_ + cg * 8;
        float4 f0 = *(const float4*)(erow);
        float4 f1 = *(const float4*)(erow + 4);
        v4u p;
        p.x = pk2(f0.x, f0.y); p.y = pk2(f0.z, f0.w);
        p.z = pk2(f1.x, f1.y); p.w = pk2(f1.z, f1.w);
        *(v4u*)&a_sh[row][cg * 8] = p;
    }
    __syncthreads();

    const int lane = tid & 31;
    const int wave = tid >> 5;            // 8 waves, 2 N-tiles of 16 each
    const int lm   = lane & 15;
    const int hi   = (lane >= 16);
    const int aoff = hi ? 16 : 0;         // bf16 A-matrix lane K offset
    const int boff = hi ? 16 : 0;         // bf16 B-matrix lane K offset

    v8f acc[2];
#pragma unroll
    for (int j = 0; j < 2; ++j) {
        float bs = bias_sum[(2 * wave + j) * 16 + lm];
#pragma unroll
        for (int r = 0; r < 8; ++r) acc[j][r] = bs;
    }

#pragma unroll
    for (int kc = 0; kc < EMB_ / 32; ++kc) {
        AB32 a;
        a.u[0] = *(const v4u*)&a_sh[lm][kc * 32 + aoff];
        a.u[1] = *(const v4u*)&a_sh[lm][kc * 32 + aoff + 8];
#pragma unroll
        for (int j = 0; j < 2; ++j) {
            int nrow = (2 * wave + j) * 16 + lm;
            const v4u* bp0 = (const v4u*)(wih_bf + (size_t)nrow * EMB_ + kc * 32 + boff);
            AB32 b; b.u[0] = bp0[0]; b.u[1] = bp0[1];
            acc[j] = wmma_bf16(a, b, acc[j]);
        }
    }

    // Store D tiles, time-major: pre[t][b][h]
#pragma unroll
    for (int j = 0; j < 2; ++j) {
        int col = (2 * wave + j) * 16 + lm;
#pragma unroll
        for (int r = 0; r < 8; ++r) {
            int m = r + (hi ? 8 : 0);
            pre[((size_t)t * B_ + btile * 16 + m) * HID_ + col] = acc[j][r];
        }
    }
}

// ---------------------------------------------------------------------------
// Sequential RNN scan + output projection + LayerNorm.
// 16 blocks: each owns a 16-row batch tile for all 512 steps.
// W_hh bf16 B-fragments are register-resident for the whole scan; the per-step
// pre[t] tile (16KB) is streamed one step ahead into an LDS double buffer via
// async-to-LDS copies (ASYNCcnt).
// ---------------------------------------------------------------------------
__global__ void __launch_bounds__(256, 1)
rnn_scan_kernel(const float* __restrict__ pre,
                const unsigned short* __restrict__ whh_bf,
                const unsigned short* __restrict__ wp_bf,
                const float* __restrict__ bpv,
                const float* __restrict__ gamma,
                const float* __restrict__ beta,
                float* __restrict__ out) {
    __shared__ __align__(16) float pre_lds[2][16 * 256];       // 2 x 16KB tiles
    __shared__ __align__(16) unsigned short hbuf[2][16][264];  // h double buffer (bf16)
    __shared__ float yv[16][128];
    __shared__ float red_s[16][16], red_q[16][16];
    __shared__ float mu_s[16], rs_s[16];

    const int btile = blockIdx.x;
    const int tid   = threadIdx.x;
    const int lane  = tid & 31;
    const int wave  = tid >> 5;
    const int lm    = lane & 15;
    const int hi    = (lane >= 16);
    const int aoff  = hi ? 16 : 0;
    const int boff  = hi ? 16 : 0;
    const int hi8   = hi ? 8 : 0;

    // h0 = 0
    {
        unsigned short* hz = &hbuf[0][0][0];
        for (int i = tid; i < 16 * 264; i += 256) hz[i] = 0;
    }

    // Register-cache W_hh B-fragments: 2 ntiles x 8 kchunks (loop-invariant).
    AB32 Bc[2][8];
#pragma unroll
    for (int j = 0; j < 2; ++j) {
        int nrow = (2 * wave + j) * 16 + lm;
        const unsigned short* wr = whh_bf + (size_t)nrow * HID_;
#pragma unroll
        for (int kc = 0; kc < 8; ++kc) {
            const v4u* p = (const v4u*)(wr + kc * 32 + boff);
            Bc[j][kc].u[0] = p[0];
            Bc[j][kc].u[1] = p[1];
        }
    }

    // Kick off the async copy of tile 0.
    const char* tile0 = (const char*)(pre + ((size_t)btile * 16) * HID_) + tid * 16;
    issue_tile_copy_async(tile0, lds_off32(&pre_lds[0][0]) + tid * 16);

    int cur = 0;
    for (int t = 0; t < T_; ++t) {
        // Stream tile t+1 into the other LDS buffer while we compute tile t.
        if (t + 1 < T_) {
            const char* g = (const char*)(pre + ((size_t)(t + 1) * B_ + btile * 16) * HID_)
                            + tid * 16;
            issue_tile_copy_async(g, lds_off32(&pre_lds[(t + 1) & 1][0]) + tid * 16);
            // 4 loads of tile t+1 may stay in flight; tile t's 4 (issued
            // earlier, completing in order) must be done.
            asm volatile("s_wait_asynccnt 0x4" ::: "memory");
        } else {
            asm volatile("s_wait_asynccnt 0x0" ::: "memory");
        }
        __syncthreads();   // all waves' copies of tile t have landed

        const float* pb = &pre_lds[t & 1][0];
        v8f acc[2];
#pragma unroll
        for (int j = 0; j < 2; ++j) {
            int col = (2 * wave + j) * 16 + lm;
#pragma unroll
            for (int r = 0; r < 8; ++r)
                acc[j][r] = pb[(r + hi8) * HID_ + col];
        }

#pragma unroll
        for (int kc = 0; kc < 8; ++kc) {
            AB32 a;
            a.u[0] = *(const v4u*)&hbuf[cur][lm][kc * 32 + aoff];
            a.u[1] = *(const v4u*)&hbuf[cur][lm][kc * 32 + aoff + 8];
            acc[0] = wmma_bf16(a, Bc[0][kc], acc[0]);
            acc[1] = wmma_bf16(a, Bc[1][kc], acc[1]);
        }

        // h_next = tanh(acc) -> LDS (bf16)
        int nxtb = cur ^ 1;
#pragma unroll
        for (int j = 0; j < 2; ++j) {
            int col = (2 * wave + j) * 16 + lm;
#pragma unroll
            for (int r = 0; r < 8; ++r)
                hbuf[nxtb][r + hi8][col] = f2bf(tanhf(acc[j][r]));
        }
        __syncthreads();   // h_next visible; everyone done reading pre_lds[t&1]
        cur = nxtb;
    }

    // ---- Output projection: y[16,128] = hT @ Wp^T + bp (wave w -> 16 cols) ----
    {
        v8f accy;
        float bc = bpv[wave * 16 + lm];
#pragma unroll
        for (int r = 0; r < 8; ++r) accy[r] = bc;

        const unsigned short* wr = wp_bf + (size_t)(wave * 16 + lm) * HID_;
#pragma unroll
        for (int kc = 0; kc < 8; ++kc) {
            AB32 a;
            a.u[0] = *(const v4u*)&hbuf[cur][lm][kc * 32 + aoff];
            a.u[1] = *(const v4u*)&hbuf[cur][lm][kc * 32 + aoff + 8];
            const v4u* p = (const v4u*)(wr + kc * 32 + boff);
            AB32 b; b.u[0] = p[0]; b.u[1] = p[1];
            accy = wmma_bf16(a, b, accy);
        }
#pragma unroll
        for (int r = 0; r < 8; ++r)
            yv[r + hi8][wave * 16 + lm] = accy[r];
    }
    __syncthreads();

    // ---- LayerNorm over OUT=128 per row ----
    {
        int r = tid >> 4, cg = tid & 15;
        float s = 0.f, q = 0.f;
#pragma unroll
        for (int c = 0; c < 8; ++c) {
            float v = yv[r][cg * 8 + c];
            s += v; q += v * v;
        }
        red_s[r][cg] = s; red_q[r][cg] = q;
        __syncthreads();
        if (cg == 0) {
            float S = 0.f, Q = 0.f;
#pragma unroll
            for (int i = 0; i < 16; ++i) { S += red_s[r][i]; Q += red_q[r][i]; }
            float mu  = S * (1.0f / OUT_);
            float var = Q * (1.0f / OUT_) - mu * mu;
            mu_s[r] = mu;
            rs_s[r] = rsqrtf(var + 1e-5f);
        }
        __syncthreads();
        float mu = mu_s[r], rs = rs_s[r];
#pragma unroll
        for (int c = 0; c < 8; ++c) {
            int col = cg * 8 + c;
            float v = (yv[r][col] - mu) * rs * gamma[col] + beta[col];
            out[((size_t)btile * 16 + r) * OUT_ + col] = v;
        }
    }
}

// ---------------------------------------------------------------------------
extern "C" void kernel_launch(void* const* d_in, const int* in_sizes, int n_in,
                              void* d_out, int out_size, void* d_ws, size_t ws_size,
                              hipStream_t stream) {
    const int*   x     = (const int*)  d_in[0];
    const float* emb   = (const float*)d_in[1];
    const float* W_ih  = (const float*)d_in[2];
    const float* W_hh  = (const float*)d_in[3];
    const float* b_ih  = (const float*)d_in[4];
    const float* b_hh  = (const float*)d_in[5];
    const float* Wp    = (const float*)d_in[6];
    const float* bp    = (const float*)d_in[7];
    const float* gamma = (const float*)d_in[8];
    const float* beta  = (const float*)d_in[9];
    float* out = (float*)d_out;

    // Workspace layout
    char* ws = (char*)d_ws;
    const size_t PRE_BYTES = (size_t)T_ * B_ * HID_ * sizeof(float);   // 128 MB
    float*          pre    = (float*)ws;
    unsigned short* wih_bf = (unsigned short*)(ws + PRE_BYTES);
    unsigned short* whh_bf = wih_bf + (size_t)HID_ * EMB_;
    unsigned short* wp_bf  = whh_bf + (size_t)HID_ * HID_;
    float*          bias_s = (float*)(wp_bf + (size_t)OUT_ * HID_);

    rnn_prep_kernel<<<128, 256, 0, stream>>>(W_ih, W_hh, b_ih, b_hh, Wp,
                                             wih_bf, whh_bf, wp_bf, bias_s);

    embed_gemm_kernel<<<dim3(B_ / 16, T_), 256, 0, stream>>>(x, emb, wih_bf,
                                                             bias_s, pre);

    rnn_scan_kernel<<<B_ / 16, 256, 0, stream>>>(pre, whh_bf, wp_bf,
                                                 bp, gamma, beta, out);
}
